// Attention1_78305843741379
// MI455X (gfx1250) — matrix operand order repaired and checked by reading, not compiled, not run on hardware
//
#include <hip/hip_runtime.h>
#include <hip/hip_bf16.h>

// ---------------- problem constants ----------------
#define B_    16
#define DIM_  256
#define M_    3072
#define HW_   4096
#define HWP_  4224   // 4096 + 64px zero pad row top & bottom
#define HEADS 8

typedef __attribute__((ext_vector_type(16))) __bf16 v16bf;
typedef __attribute__((ext_vector_type(8)))  float  v8f;
typedef __attribute__((ext_vector_type(2)))  float  v2f;
typedef __attribute__((ext_vector_type(4)))  unsigned int v4u;
typedef __attribute__((ext_vector_type(4)))  int    v4i;
typedef __attribute__((ext_vector_type(8)))  int    v8i;

#if defined(__has_builtin)
#if __has_builtin(__builtin_amdgcn_tensor_load_to_lds)
#define HAVE_TDM 1
#else
#define HAVE_TDM 0
#warning "CDNA5 probe: no __builtin_amdgcn_tensor_load_to_lds - cooperative LDS fill fallback"
#endif
#if !__has_builtin(__builtin_amdgcn_wmma_f32_16x16x4_f32)
#warning "CDNA5 probe: no __builtin_amdgcn_wmma_f32_16x16x4_f32 - scalar fallback in attention"
#endif
#else
#define HAVE_TDM 0
#endif

struct Frag { union { v16bf v; uint4 q[2]; unsigned u[8]; }; };

// A-matrix 16x32 bf16 fragment (row-major, K-contiguous source).
// lanes 0-15: K 0..15 ; lanes 16-31: K 16..31
__device__ __forceinline__ v16bf load_frag_A(const __bf16* rowp, int half) {
  Frag f;
  f.q[0] = *(const uint4*)(rowp + 16 * half);
  f.q[1] = *(const uint4*)(rowp + 16 * half + 8);
  return f.v;
}

// B-matrix 32x16 bf16 fragment from a K-contiguous column.
// lanes 0-15: K 0..15 ; lanes 16-31: K 16..31
__device__ __forceinline__ v16bf load_frag_B(const __bf16* colp, int half) {
  Frag f;
  f.q[0] = *(const uint4*)(colp + 16 * half);
  f.q[1] = *(const uint4*)(colp + 16 * half + 8);
  return f.v;
}

__device__ __forceinline__ v8f wmma_bf16(v16bf a, v16bf b, v8f c) {
  return __builtin_amdgcn_wmma_f32_16x16x32_bf16(false, a, false, b, (short)0, c,
                                                 false, false);
}

// fp32 WMMA (16x16x4) for the precision-sensitive attention math
__device__ __forceinline__ v8f wmma_f32x4(float a0, float a1, float b0, float b1,
                                          v8f c) {
#if __has_builtin(__builtin_amdgcn_wmma_f32_16x16x4_f32)
  v2f a = {a0, a1};
  v2f b = {b0, b1};
  return __builtin_amdgcn_wmma_f32_16x16x4_f32(false, a, false, b, (short)0, c,
                                               false, false);
#else
  c[0] += a0 * b0 + a1 * b1;  // compile-safe fallback only
  return c;
#endif
}

// ---- Tensor Data Mover: 2-D tile (rowElems x rows, 2-byte elems) -> LDS ----
// D# per CDNA5 ISA: group0 = {flags, lds_addr,
// global_addr, type=2}; group1 = {data_size=2B, tensor/tile dims, stride}.
__device__ __forceinline__ void tdm_load_2d(unsigned lds_addr, const void* gaddr,
                                            unsigned rowElems, unsigned rows,
                                            unsigned strideElems) {
#if HAVE_TDM
  unsigned long long ga = (unsigned long long)(uintptr_t)gaddr;
  v4u g0;
  g0[0] = 1u;                                   // count=1 (valid), user mode
  g0[1] = lds_addr;                             // lds_addr (bytes)
  g0[2] = (unsigned)(ga & 0xffffffffu);         // global_addr[31:0]
  g0[3] = (unsigned)((ga >> 32) & 0x01ffffffu)  // global_addr[56:32]
          | (2u << 30);                         // type=2 ("image")
  v8i g1;
  g1[0] = (int)(1u << 16);                      // data_size=1 -> 2 bytes
  g1[1] = (int)((rowElems & 0xffffu) << 16);    // tensor_dim0[15:0]
  g1[2] = (int)(((rowElems >> 16) & 0xffffu) |  // tensor_dim0[31:16]
                ((rows & 0xffffu) << 16));      // tensor_dim1[15:0]
  g1[3] = (int)(((rows >> 16) & 0xffffu) |      // tensor_dim1[31:16]
                ((rowElems & 0xffffu) << 16));  // tile_dim0
  g1[4] = (int)(rows & 0xffffu);                // tile_dim1 ; tile_dim2=0 (2D)
  g1[5] = (int)strideElems;                     // tensor_dim0_stride[31:0]
  g1[6] = 0;
  g1[7] = 0;
  v4i z = {0, 0, 0, 0};
#if __clang_major__ >= 23
  v8i z8 = {0, 0, 0, 0, 0, 0, 0, 0};
  __builtin_amdgcn_tensor_load_to_lds(g0, g1, z, z, z8, 0);
#else
  __builtin_amdgcn_tensor_load_to_lds(g0, g1, z, z, 0);
#endif
#endif
}

// ---------------- casts / packing ----------------
__global__ __launch_bounds__(256) void k_cast(const float* __restrict__ s,
                                              __bf16* __restrict__ d, int n) {
  int i = blockIdx.x * 256 + threadIdx.x;
  if (i < n) d[i] = (__bf16)s[i];
}

__global__ __launch_bounds__(256) void k_zero_bf16(__bf16* __restrict__ d,
                                                   int n) {
  int i = blockIdx.x * 256 + threadIdx.x;
  if (i < n) d[i] = (__bf16)0.0f;
}

// (B, rows, cols) f32 -> (B, cols, rows) bf16   (channel-last, K-contiguous)
__global__ __launch_bounds__(256) void k_castT(const float* __restrict__ s,
                                               __bf16* __restrict__ d, int rows,
                                               int cols) {
  long i = (long)blockIdx.x * 256 + threadIdx.x;
  long rc = (long)rows * cols;
  int b = (int)(i / rc);
  int rem = (int)(i - (long)b * rc);
  int r = rem / cols, c = rem - r * cols;
  d[((long)b * cols + c) * rows + r] = (__bf16)s[i];
}

// W_dw (O,I,3,3) f32 -> (t,O,I) bf16
__global__ __launch_bounds__(256) void k_pack_wdw(const float* __restrict__ s,
                                                  __bf16* __restrict__ d) {
  int i = blockIdx.x * 256 + threadIdx.x;  // < 256*256*9
  int o = i / (256 * 9);
  int rem = i - o * (256 * 9);
  int ic = rem / 9, t = rem - ic * 9;
  d[(long)t * 65536 + (long)o * 256 + ic] = (__bf16)s[i];
}

// ---------------- generic bf16 WMMA GEMM, TDM-staged A & B tiles -----------
// D(MxN) = A(MxK) * B(KxN). A row-major (ldA), B pixel-major: elem(k,n) at
// B[n*ldB + k], K == 256. OM: 0 = f32 (M,N); 1 = f32 (N,M); 2 = bf16 (N,M).
template <int OM>
__global__ __launch_bounds__(256) void k_gemm_bf16(
    const __bf16* __restrict__ A, int ldA, const __bf16* __restrict__ Bm,
    int ldB, long sBb, void* __restrict__ Dp, int ldD, long sDb, int K) {
  __shared__ __bf16 ldsA[64 * 256];  // 32 KB: 64-row x K=256 A tile
  __shared__ __bf16 ldsB[64 * 256];  // 32 KB: 64-row x K=256 B tile
  const int b = blockIdx.z;
  const int wave = threadIdx.x >> 5, lane = threadIdx.x & 31;
  const int half = lane >> 4, l15 = lane & 15;
  const int m0 = blockIdx.y * 64 + (wave >> 1) * 16;
  const int n0 = blockIdx.x * 64 + (wave & 1) * 32;

  const __bf16* ga = A + (long)(blockIdx.y * 64) * ldA;
  const __bf16* gb = Bm + (long)b * sBb + (long)(blockIdx.x * 64) * ldB;
#if HAVE_TDM
  if (wave == 0) {
    tdm_load_2d((unsigned)(uintptr_t)ldsA, ga, 256, 64, (unsigned)ldA);
    tdm_load_2d((unsigned)(uintptr_t)ldsB, gb, 256, 64, (unsigned)ldB);
    __builtin_amdgcn_s_wait_tensorcnt(0);
  }
#else
  for (int i = threadIdx.x; i < 1024; i += 256) {
    *(uint4*)(ldsA + (i >> 5) * 256 + (i & 31) * 8) =
        *(const uint4*)(ga + (long)(i >> 5) * ldA + (i & 31) * 8);
    *(uint4*)(ldsB + (i >> 5) * 256 + (i & 31) * 8) =
        *(const uint4*)(gb + (long)(i >> 5) * ldB + (i & 31) * 8);
  }
#endif
  __syncthreads();

  v8f acc0 = {}, acc1 = {};
  const __bf16* arow = ldsA + ((wave >> 1) * 16 + l15) * 256;
  const int nl = (wave & 1) * 32 + l15;       // local B column in LDS
  const __bf16* brow0 = ldsB + nl * 256;
  const __bf16* brow1 = ldsB + (nl + 16) * 256;
  for (int k0 = 0; k0 < K; k0 += 32) {
    v16bf a = load_frag_A(arow + k0, half);    // ds_load_b128 x2
    v16bf b0 = load_frag_B(brow0 + k0, half);  // ds_load_b128 x2
    v16bf b1 = load_frag_B(brow1 + k0, half);
    acc0 = wmma_bf16(a, b0, acc0);
    acc1 = wmma_bf16(a, b1, acc1);
  }
  const int col0 = n0 + l15, col1 = col0 + 16;
  if constexpr (OM == 0) {
    float* D = (float*)Dp + (long)b * sDb;
    for (int r = 0; r < 8; r++) {
      int mm = m0 + r + 8 * half;
      D[(long)mm * ldD + col0] = acc0[r];
      D[(long)mm * ldD + col1] = acc1[r];
    }
  } else if constexpr (OM == 1) {
    float* D = (float*)Dp + (long)b * sDb;
    for (int r = 0; r < 8; r++) {
      int mm = m0 + r + 8 * half;
      D[(long)col0 * ldD + mm] = acc0[r];
      D[(long)col1 * ldD + mm] = acc1[r];
    }
  } else {
    __bf16* D = (__bf16*)Dp + (long)b * sDb;
    for (int r = 0; r < 8; r++) {
      int mm = m0 + r + 8 * half;
      D[(long)col0 * ldD + mm] = (__bf16)acc0[r];
      D[(long)col1 * ldD + mm] = (__bf16)acc1[r];
    }
  }
}

// ---------------- 3x3 conv as implicit GEMM, double-buffered TDM -----------
// Q0p is (B, HWP_, 256) with one zeroed 64-px pad row top & bottom, so the
// rows y-1..y+1 slice is 192 *consecutive* pixels: one 2-D TDM tile per
// 32-channel K-step (32 elems x 192 rows, stride 256 = 12 KB). Two slices
// kept in flight: issue slice i+1, wait tensorcnt<=1 for slice i (in-order).
__device__ __forceinline__ v16bf frag_lds_img(const __bf16* ldsT, int lr,
                                              int sx, int half) {
  Frag f;
  if (sx >= 0 && sx < 64) {
    const __bf16* p = ldsT + (lr * 64 + sx) * 32 + 16 * half;
    f.q[0] = *(const uint4*)p;
    f.q[1] = *(const uint4*)(p + 8);
  } else {  // x zero-pad
    f.q[0] = make_uint4(0u, 0u, 0u, 0u);
    f.q[1] = make_uint4(0u, 0u, 0u, 0u);
  }
  return f.v;
}

__global__ __launch_bounds__(256) void k_conv3x3(
    const __bf16* __restrict__ Wp,    // (9, 256, 256)
    const __bf16* __restrict__ Q0p,   // (B, HWP_, 256) padded pixel-major
    float* __restrict__ Qc) {         // (B, 4096, 256)
  __shared__ __bf16 ldsT[2][192 * 32];  // 2 x 12 KB ping-pong slices
  const int b = blockIdx.z, y = blockIdx.y, ot = blockIdx.x;
  const int wave = threadIdx.x >> 5, lane = threadIdx.x & 31;
  const int half = lane >> 4, l15 = lane & 15;
  const int o0 = ot * 64 + (wave >> 1) * 16;
  const int p0 = (wave & 1) * 32;
  // padded pixel of row (y-1) start = y*64
  const __bf16* gbase = Q0p + ((long)b * HWP_ + (long)y * 64) * 256;
  v8f acc0 = {}, acc1 = {};
#if HAVE_TDM
  if (wave == 0)
    tdm_load_2d((unsigned)(uintptr_t)&ldsT[0][0], gbase, 32, 192, 256);
#endif
  for (int i = 0; i < 8; i++) {          // K-step: channels cb = i*32
    const int cb = i * 32;
    __syncthreads();  // ping-pong buffer (i+1)&1 fully consumed
#if HAVE_TDM
    if (wave == 0) {
      if (i + 1 < 8) {
        tdm_load_2d((unsigned)(uintptr_t)&ldsT[(i + 1) & 1][0],
                    gbase + cb + 32, 32, 192, 256);
        __builtin_amdgcn_s_wait_tensorcnt(1);  // slice i done (in-order)
      } else {
        __builtin_amdgcn_s_wait_tensorcnt(0);
      }
    }
#else
    for (int j = threadIdx.x; j < 768; j += 256)
      *(uint4*)(&ldsT[i & 1][0] + (j >> 2) * 32 + (j & 3) * 8) =
          *(const uint4*)(gbase + cb + (long)(j >> 2) * 256 + (j & 3) * 8);
#endif
    __syncthreads();
    const __bf16* lsl = &ldsT[i & 1][0];
    for (int t = 0; t < 9; t++) {
      int lr = t / 3, kx = t % 3 - 1;  // local row 0..2 = y-1..y+1
      const __bf16* arow = Wp + (long)t * 65536 + (long)(o0 + l15) * 256 + cb;
      v16bf a = load_frag_A(arow, half);
      int sx0 = p0 + l15 + kx;
      v16bf b0 = frag_lds_img(lsl, lr, sx0, half);
      v16bf b1 = frag_lds_img(lsl, lr, sx0 + 16, half);
      acc0 = wmma_bf16(a, b0, acc0);
      acc1 = wmma_bf16(a, b1, acc1);
    }
  }
  int px0 = p0 + l15;
  float* dst = Qc + ((long)b * HW_ + y * 64) * 256;
  for (int r = 0; r < 8; r++) {
    int oc = o0 + r + 8 * half;
    dst[(long)px0 * 256 + oc] = acc0[r];
    dst[(long)(px0 + 16) * 256 + oc] = acc1[r];
  }
}

// ---------------- snake reorder + linear interpolation 4096 -> 3072 --------
__global__ __launch_bounds__(256) void k_snake_interp(
    const float* __restrict__ qc, float* __restrict__ qi) {
  int b = blockIdx.y, mi = blockIdx.x, ch = threadIdx.x;
  float src = (mi + 0.5f) * (4096.0f / 3072.0f) - 0.5f;
  src = fmaxf(src, 0.0f);
  int i0 = (int)floorf(src);
  if (i0 > 4095) i0 = 4095;
  int i1 = i0 + 1;
  if (i1 > 4095) i1 = 4095;
  float lam = src - (float)i0;
  int r0 = i0 >> 6, c0 = i0 & 63;
  int p0 = (r0 & 1) ? (r0 << 6) + (63 - c0) : i0;
  int r1 = i1 >> 6, c1 = i1 & 63;
  int p1 = (r1 & 1) ? (r1 << 6) + (63 - c1) : i1;
  const float* base = qc + (long)b * HW_ * DIM_;
  float v0 = base[(long)p0 * DIM_ + ch];
  float v1 = base[(long)p1 * DIM_ + ch];
  qi[((long)b * M_ + mi) * DIM_ + ch] = v0 * (1.0f - lam) + v1 * lam;
}

// ---------------- fused per-(b,h) channel attention (fp32 WMMA) ------------
__global__ __launch_bounds__(256) void k_attention(
    const float* __restrict__ qi, const float* __restrict__ kvp,
    const float* __restrict__ temp, __bf16* __restrict__ ao) {
  __shared__ float Sm[32][33];
  __shared__ float nq[32];
  __shared__ float nk[32];
  const int h = blockIdx.x, b = blockIdx.y;
  const int h32 = h * 32;
  const int tid = threadIdx.x;
  const int wave = tid >> 5, lane = tid & 31;
  const int half = lane >> 4, l15 = lane & 15;
  const long qbase = (long)b * M_ * 256;
  const long kbase = (long)b * M_ * 512;

  if (tid < 32) { nq[tid] = 0.0f; nk[tid] = 0.0f; }
  __syncthreads();
  {
    int rowc = tid >> 3, sub = tid & 7;
    float sq = 0.0f, sk = 0.0f;
    for (int m = sub; m < M_; m += 8) {
      float qv = qi[qbase + (long)m * 256 + h32 + rowc];
      float kv = kvp[kbase + (long)m * 512 + h32 + rowc];
      sq += qv * qv;
      sk += kv * kv;
    }
    atomicAdd(&nq[rowc], sq);  // ds_add_f32
    atomicAdd(&nk[rowc], sk);
  }
  __syncthreads();
  if (tid < 32) {
    nq[tid] = fmaxf(sqrtf(nq[tid]), 1e-12f);
    nk[tid] = fmaxf(sqrtf(nk[tid]), 1e-12f);
  }
  __syncthreads();

  if (wave < 4) {  // S = q k^T (32x32, K=3072)
    const int ct = wave & 1, dt = wave >> 1;
    v8f acc = {};
    const int koff = 2 * half;
    for (int k0 = 0; k0 < M_; k0 += 4) {
      long q0 = qbase + (long)(k0 + koff) * 256 + h32 + ct * 16 + l15;
      long k0i = kbase + (long)(k0 + koff) * 512 + h32 + dt * 16 + l15;
      float a0 = qi[q0];
      float a1 = qi[q0 + 256];
      float b0 = kvp[k0i];
      float b1 = kvp[k0i + 512];
      acc = wmma_f32x4(a0, a1, b0, b1, acc);
    }
    float tf = temp[h];
    for (int r = 0; r < 8; r++) {
      int cc = ct * 16 + r + 8 * half;
      int dd = dt * 16 + l15;
      Sm[cc][dd] = acc[r] * tf / (nq[cc] * nk[dd]);
    }
  }
  __syncthreads();

  if (tid < 32) {  // row softmax over 32
    float mx = -3.4e38f;
    for (int d = 0; d < 32; d++) mx = fmaxf(mx, Sm[tid][d]);
    float s = 0.0f;
    for (int d = 0; d < 32; d++) {
      float e = __expf(Sm[tid][d] - mx);
      Sm[tid][d] = e;
      s += e;
    }
    float inv = 1.0f / s;
    for (int d = 0; d < 32; d++) Sm[tid][d] *= inv;
  }
  __syncthreads();

  const int koff = 2 * half;  // O = S * v (K=32, N=3072)
  for (int idx = wave; idx < 384; idx += 8) {
    const int ct = idx & 1;
    const int n0 = (idx >> 1) * 16;
    v8f acc = {};
    for (int k0 = 0; k0 < 32; k0 += 4) {
      float a0 = Sm[ct * 16 + l15][k0 + koff];
      float a1 = Sm[ct * 16 + l15][k0 + koff + 1];
      long vb = kbase + (long)(n0 + l15) * 512 + 256 + h32 + k0 + koff;
      float b0 = kvp[vb];
      float b1 = kvp[vb + 1];
      acc = wmma_f32x4(a0, a1, b0, b1, acc);
    }
    for (int r = 0; r < 8; r++) {
      int cc = ct * 16 + r + 8 * half;
      ao[((long)b * M_ + n0 + l15) * 256 + h32 + cc] = (__bf16)acc[r];
    }
  }
}

// ---------------- host launcher ----------------
extern "C" void kernel_launch(void* const* d_in, const int* in_sizes, int n_in,
                              void* d_out, int out_size, void* d_ws,
                              size_t ws_size, hipStream_t stream) {
  (void)in_sizes; (void)n_in; (void)out_size; (void)ws_size;
  const float* x = (const float*)d_in[0];
  const float* y = (const float*)d_in[1];
  const float* Wkv = (const float*)d_in[2];
  const float* Wq = (const float*)d_in[3];
  const float* Wdw = (const float*)d_in[4];
  const float* Wpo = (const float*)d_in[5];
  const float* temp = (const float*)d_in[6];

  char* ws = (char*)d_ws;
  size_t off = 0;
  auto take = [&](size_t bytes) -> char* {
    char* p = ws + off;
    off = (off + bytes + 255) & ~(size_t)255;
    return p;
  };
  __bf16* xt   = (__bf16*)take((size_t)B_ * M_ * DIM_ * 2);    // (B, M, 256)
  __bf16* yt   = (__bf16*)take((size_t)B_ * HW_ * DIM_ * 2);   // (B, 4096, 256)
  __bf16* wkvb = (__bf16*)take((size_t)512 * 256 * 2);
  __bf16* wqb  = (__bf16*)take((size_t)256 * 256 * 2);
  __bf16* wdwp = (__bf16*)take((size_t)9 * 256 * 256 * 2);     // (t, O, I)
  __bf16* wpob = (__bf16*)take((size_t)256 * 256 * 2);
  float*  kv   = (float*)take((size_t)B_ * M_ * 512 * 4);      // (B, M, 512)
  __bf16* q0p  = (__bf16*)take((size_t)B_ * HWP_ * DIM_ * 2);  // padded
  float*  qc   = (float*)take((size_t)B_ * HW_ * DIM_ * 4);    // (B, 4096, 256)
  float*  qiw  = (float*)take((size_t)B_ * M_ * DIM_ * 4);     // (B, M, 256)
  __bf16* ao   = (__bf16*)take((size_t)B_ * M_ * DIM_ * 2);    // (B, M, 256)
  float*  out  = (float*)d_out;                                // (B, 256, M)

  // 1) casts / packing into WMMA-friendly (K-contiguous) layouts
  k_castT<<<dim3((B_ * DIM_ * M_) / 256), 256, 0, stream>>>(x, xt, DIM_, M_);
  k_castT<<<dim3((B_ * DIM_ * HW_) / 256), 256, 0, stream>>>(y, yt, DIM_, HW_);
  k_cast<<<dim3((512 * 256) / 256), 256, 0, stream>>>(Wkv, wkvb, 512 * 256);
  k_cast<<<dim3((256 * 256) / 256), 256, 0, stream>>>(Wq, wqb, 256 * 256);
  k_cast<<<dim3((256 * 256) / 256), 256, 0, stream>>>(Wpo, wpob, 256 * 256);
  k_pack_wdw<<<dim3((9 * 256 * 256) / 256), 256, 0, stream>>>(Wdw, wdwp);
  k_zero_bf16<<<dim3((B_ * HWP_ * DIM_) / 256), 256, 0, stream>>>(
      q0p, B_ * HWP_ * DIM_);

  // 2) kv = W_kv @ x -> (B, M, 512) f32
  k_gemm_bf16<1><<<dim3(M_ / 64, 512 / 64, B_), 256, 0, stream>>>(
      wkvb, 256, xt, 256, (long)M_ * 256, kv, 512, (long)M_ * 512, 256);

  // 3) q0 = W_q @ y (1x1) -> interior of padded (B, HWP_, 256) bf16
  k_gemm_bf16<2><<<dim3(HW_ / 64, 256 / 64, B_), 256, 0, stream>>>(
      wqb, 256, yt, 256, (long)HW_ * 256, q0p + (size_t)64 * 256, 256,
      (long)HWP_ * 256, 256);

  // 4) qc = conv3x3(q0) -> (B, 4096, 256) f32 (implicit GEMM, 77 GFLOP)
  k_conv3x3<<<dim3(4, 64, B_), 256, 0, stream>>>(wdwp, q0p, qc);

  // 5) snake reorder + linear interp -> (B, M, 256) f32
  k_snake_interp<<<dim3(M_, B_), 256, 0, stream>>>(qc, qiw);

  // 6) fused l2norm + qk^T + softmax + S@v per (b, h) -> (B, M, 256) bf16
  k_attention<<<dim3(HEADS, B_), 256, 0, stream>>>(qiw, kv, temp, ao);

  // 7) out = W_po @ attn_out -> (B, 256, M) f32 in reference layout
  k_gemm_bf16<0><<<dim3(M_ / 64, 256 / 64, B_), 256, 0, stream>>>(
      wpob, 256, ao, 256, (long)M_ * 256, out, M_, (long)256 * M_, 256);
}
